// EdgeProbGCN_48155173323171
// MI455X (gfx1250) — matrix operand order repaired and checked by reading, not compiled, not run on hardware
//
#include <hip/hip_runtime.h>
#include <hip/hip_bf16.h>

typedef __attribute__((ext_vector_type(16))) _Float16 v16h;
typedef __attribute__((ext_vector_type(8)))  _Float16 h8;
typedef __attribute__((ext_vector_type(8)))  float    v8f;

#define A_LD  136   // 128 + 8 halves pad: dword stride 68 -> bank stride 4, conflict-free
#define EF_LD 264   // 256 + 8 halves pad: dword stride 132 -> bank stride 4, conflict-free

// ---------------------------------------------------------------------------
// Small utility kernels
// ---------------------------------------------------------------------------
__global__ void k_zero_f32(float* __restrict__ p, int n) {
    int i = blockIdx.x * blockDim.x + threadIdx.x;
    if (i < n) p[i] = 0.0f;
}

__global__ void k_deg_count(const int* __restrict__ dst, float* __restrict__ deg, int E) {
    int e = blockIdx.x * blockDim.x + threadIdx.x;
    if (e < E) atomicAdd(&deg[dst[e]], 1.0f);
}

__global__ void k_deg_final(const float* __restrict__ deg,
                            float* __restrict__ dinv,
                            float* __restrict__ invd, int N) {
    int i = blockIdx.x * blockDim.x + threadIdx.x;
    if (i < N) {
        float dg = deg[i] + 1.0f;        // +1 for self loop
        dinv[i] = rsqrtf(dg);
        invd[i] = 1.0f / dg;
    }
}

__global__ void k_f32_to_f16(const float* __restrict__ in, _Float16* __restrict__ out, int n) {
    int i = blockIdx.x * blockDim.x + threadIdx.x;
    if (i < n) out[i] = (_Float16)in[i];
}

// in: [K][Nc] row-major f32  ->  out: [Nc][K] f16 (n-major, contiguous K per row)
__global__ void k_transpose_f16(const float* __restrict__ in, _Float16* __restrict__ out,
                                int K, int Nc) {
    int i = blockIdx.x * blockDim.x + threadIdx.x;
    if (i < K * Nc) {
        int k = i / Nc, n = i - k * Nc;
        out[(size_t)n * K + k] = (_Float16)in[i];
    }
}

// ---------------------------------------------------------------------------
// Fragment helpers (per ISA 16-bit A 16x32 / B 32x16 VGPR layouts)
// ---------------------------------------------------------------------------
union FragU { v16h v; unsigned u[8]; };

__device__ __forceinline__ v16h load_a_frag(const _Float16* pRow, int kb, int half) {
    FragU f;
#pragma unroll
    for (int p = 0; p < 8; ++p) {
        int ka = kb + ((p >= 4) ? 16 : 0) + half * 8 + (p & 3) * 2;
        f.u[p] = *(const unsigned*)(pRow + ka);
    }
    return f.v;
}

__device__ __forceinline__ v16h load_b_frag(const _Float16* pRow, int kb, int half) {
    FragU f;
#pragma unroll
    for (int p = 0; p < 8; ++p) {
        int kw = kb + ((p >= 4) ? 16 : 0) + half * 8 + (p & 3) * 2;
        f.u[p] = *(const unsigned*)(pRow + kw);
    }
    return f.v;
}

// ---------------------------------------------------------------------------
// Node GEMM: C[N][128] = A_f16[N][128] @ W (Wt is [128 n][128 k] f16)
// 256 threads = 8 waves; block covers 32 rows; wave w -> cols w*16..w*16+15,
// two 16-row tiles sharing each B fragment.
// ---------------------------------------------------------------------------
__global__ void k_gemm_node(const _Float16* __restrict__ Af16,
                            const _Float16* __restrict__ Wt,
                            float* __restrict__ Cout, int N) {
    __shared__ _Float16 As[32 * A_LD];

    int t    = threadIdx.x;
    int row0 = blockIdx.x * 32;

    // stage 32x128 A tile into LDS (thread t -> row t/8, 16-elem segment t%8)
    {
        int r  = t >> 3;
        int c0 = (t & 7) * 16;
        int m  = row0 + r; if (m >= N) m = N - 1;
        const _Float16* gp = Af16 + (size_t)m * 128 + c0;
        h8 a0 = *(const h8*)(gp);
        h8 a1 = *(const h8*)(gp + 8);
        *(h8*)&As[r * A_LD + c0]     = a0;
        *(h8*)&As[r * A_LD + c0 + 8] = a1;
    }
    __syncthreads();

    int wave = t >> 5;
    int lane = t & 31;
    int l15  = lane & 15;
    int half = lane >> 4;
    int n0   = wave * 16;

    const _Float16* pA0 = As + (size_t)l15 * A_LD;
    const _Float16* pA1 = As + (size_t)(l15 + 16) * A_LD;
    const _Float16* pB  = Wt + (size_t)(n0 + l15) * 128;

    v8f acc0 = {}, acc1 = {};
#pragma unroll
    for (int kb = 0; kb < 128; kb += 32) {
        v16h fb  = load_b_frag(pB, kb, half);
        v16h fa0 = load_a_frag(pA0, kb, half);
        v16h fa1 = load_a_frag(pA1, kb, half);
        acc0 = __builtin_amdgcn_wmma_f32_16x16x32_f16(
            false, fa0, false, fb, (short)0, acc0, false, false);
        acc1 = __builtin_amdgcn_wmma_f32_16x16x32_f16(
            false, fa1, false, fb, (short)0, acc1, false, false);
    }

    if (row0 + 32 <= N) {   // fast path: straight-line stores (N%32==0 normally)
        float* c0p = Cout + (size_t)(row0 + half * 8) * 128 + n0 + l15;
#pragma unroll
        for (int r = 0; r < 8; ++r) c0p[(size_t)r * 128]          = acc0[r];
#pragma unroll
        for (int r = 0; r < 8; ++r) c0p[(size_t)(r + 16) * 128]   = acc1[r];
    } else {
#pragma unroll
        for (int r = 0; r < 8; ++r) {
            int mm = row0 + r + half * 8;
            if (mm < N)      Cout[(size_t)mm * 128 + n0 + l15]        = acc0[r];
            if (mm + 16 < N) Cout[(size_t)(mm + 16) * 128 + n0 + l15] = acc1[r];
        }
    }
}

// ---------------------------------------------------------------------------
// Edge scatter: agg[dst] += xw[src] * dinv[src]*dinv[dst]   (32 lanes / edge)
// ---------------------------------------------------------------------------
__global__ void k_scatter(const float* __restrict__ xw,
                          const int* __restrict__ src,
                          const int* __restrict__ dst,
                          const float* __restrict__ dinv,
                          float* __restrict__ agg, int E) {
    int gid  = blockIdx.x * blockDim.x + threadIdx.x;
    int e    = gid >> 5;
    int lane = gid & 31;
    if (e >= E) return;
    int s = src[e], d = dst[e];
    float nrm = dinv[s] * dinv[d];
    int c = lane * 4;
    float4 v = *(const float4*)(xw + (size_t)s * 128 + c);
    float* ap = agg + (size_t)d * 128 + c;
    atomicAdd(ap + 0, v.x * nrm);
    atomicAdd(ap + 1, v.y * nrm);
    atomicAdd(ap + 2, v.z * nrm);
    atomicAdd(ap + 3, v.w * nrm);
}

// hf16 = relu(agg + xw * (1/deg) + b), f16 (feeds both next GEMM and edge head)
__global__ void k_node_update(const float* __restrict__ agg,
                              const float* __restrict__ xw,
                              const float* __restrict__ invd,
                              const float* __restrict__ bias,
                              _Float16* __restrict__ hf16, int N) {
    int idx = blockIdx.x * blockDim.x + threadIdx.x;
    if (idx >= N * 128) return;
    int i = idx >> 7;
    int c = idx & 127;
    float v = agg[idx] + xw[idx] * invd[i] + bias[c];
    hf16[idx] = (_Float16)fmaxf(v, 0.0f);
}

// ---------------------------------------------------------------------------
// Edge head: 32 edges / block. Build ef=[x*y | x-y] (32 x 256 f16, padded LDS),
// z = relu(ef @ fc1 + b), prob = sigmoid(z . fc2w + fc2b).
// 8 waves; wave w -> cols w*16, two 16-edge row tiles sharing B fragments.
// ---------------------------------------------------------------------------
__global__ void k_edge_head(const _Float16* __restrict__ h2,   // [N][128] f16
                            const int* __restrict__ src,
                            const int* __restrict__ dst,
                            const _Float16* __restrict__ fc1t, // [128 n][256 k] f16
                            const float* __restrict__ fc1b,
                            const float* __restrict__ fc2w,    // [128]
                            const float* __restrict__ fc2b,    // [1]
                            float* __restrict__ out, int E) {
    __shared__ _Float16 ef[32 * EF_LD];
    __shared__ float zred[32];

    int t  = threadIdx.x;
    int e0 = blockIdx.x * 32;

    // build tile: thread t -> edge i=t/8, 16-channel segment t%8; writes both
    // the product half (cols c0..) and the difference half (cols 128+c0..)
    {
        int i  = t >> 3;
        int c0 = (t & 7) * 16;
        int e  = e0 + i; if (e >= E) e = E - 1;
        int s = src[e], d = dst[e];
        const _Float16* xp = h2 + (size_t)s * 128 + c0;
        const _Float16* yp = h2 + (size_t)d * 128 + c0;
        h8 x0 = *(const h8*)(xp);      h8 x1 = *(const h8*)(xp + 8);
        h8 y0 = *(const h8*)(yp);      h8 y1 = *(const h8*)(yp + 8);
        _Float16* row = ef + (size_t)i * EF_LD;
        *(h8*)&row[c0]           = x0 * y0;
        *(h8*)&row[c0 + 8]       = x1 * y1;
        *(h8*)&row[128 + c0]     = x0 - y0;
        *(h8*)&row[128 + c0 + 8] = x1 - y1;
    }
    if (t < 32) zred[t] = 0.0f;
    __syncthreads();

    int wave = t >> 5;
    int lane = t & 31;
    int l15  = lane & 15;
    int half = lane >> 4;
    int n0   = wave * 16;

    const _Float16* pA0 = ef + (size_t)l15 * EF_LD;
    const _Float16* pA1 = ef + (size_t)(l15 + 16) * EF_LD;
    const _Float16* pB  = fc1t + (size_t)(n0 + l15) * 256;

    v8f acc0 = {}, acc1 = {};
#pragma unroll
    for (int kb = 0; kb < 256; kb += 32) {
        v16h fb  = load_b_frag(pB, kb, half);
        v16h fa0 = load_a_frag(pA0, kb, half);
        v16h fa1 = load_a_frag(pA1, kb, half);
        acc0 = __builtin_amdgcn_wmma_f32_16x16x32_f16(
            false, fa0, false, fb, (short)0, acc0, false, false);
        acc1 = __builtin_amdgcn_wmma_f32_16x16x32_f16(
            false, fa1, false, fb, (short)0, acc1, false, false);
    }

    // z = relu(acc + b1); dot with fc2w: shuffle-reduce over the wave's 16
    // cols (lanes sharing the same half), then across waves via LDS atomics.
    float w2  = fc2w[n0 + l15];
    float b1v = fc1b[n0 + l15];
#pragma unroll
    for (int r = 0; r < 8; ++r) {
        float z0 = acc0[r] + b1v; z0 = z0 > 0.0f ? z0 : 0.0f;
        float z1 = acc1[r] + b1v; z1 = z1 > 0.0f ? z1 : 0.0f;
        float v0 = z0 * w2;
        float v1 = z1 * w2;
#pragma unroll
        for (int offs = 1; offs < 16; offs <<= 1) {
            v0 += __shfl_xor(v0, offs, 32);
            v1 += __shfl_xor(v1, offs, 32);
        }
        if (l15 == 0) {
            atomicAdd(&zred[r + half * 8],      v0);
            atomicAdd(&zred[16 + r + half * 8], v1);
        }
    }
    __syncthreads();

    if (t < 32) {
        int e = e0 + t;
        if (e < E) {
            float v = zred[t] + fc2b[0];
            out[e] = 1.0f / (1.0f + __expf(-v));
        }
    }
}

// ---------------------------------------------------------------------------
// Host-side launcher
// ---------------------------------------------------------------------------
extern "C" void kernel_launch(void* const* d_in, const int* in_sizes, int n_in,
                              void* d_out, int out_size, void* d_ws, size_t ws_size,
                              hipStream_t stream) {
    (void)n_in; (void)out_size; (void)ws_size;

    const int N = in_sizes[0] / 128;   // nodes
    const int E = in_sizes[1] / 2;     // edges

    const float* x     = (const float*)d_in[0];
    const int*   src   = (const int*)d_in[1];
    const int*   dst   = src + E;
    const float* W1    = (const float*)d_in[2];
    const float* b1    = (const float*)d_in[3];
    const float* W2    = (const float*)d_in[4];
    const float* b2    = (const float*)d_in[5];
    const float* fc1w  = (const float*)d_in[6];
    const float* fc1b  = (const float*)d_in[7];
    const float* fc2w  = (const float*)d_in[8];
    const float* fc2b  = (const float*)d_in[9];
    float*       out   = (float*)d_out;

    // workspace carve-up
    size_t off = 0;
    auto take = [&](size_t bytes) -> char* {
        size_t o = (off + 255) & ~(size_t)255;
        off = o + bytes;
        return (char*)d_ws + o;
    };
    const size_t nF = (size_t)N * 128;
    _Float16* hf16 = (_Float16*)take(nF * sizeof(_Float16));  // gemm/gather input
    float*    xw   = (float*)take(nF * sizeof(float));        // gemm output
    float*    agg  = (float*)take(nF * sizeof(float));        // scatter accum
    float*    deg  = (float*)take((size_t)N * sizeof(float));
    float*    dinv = (float*)take((size_t)N * sizeof(float));
    float*    invd = (float*)take((size_t)N * sizeof(float));
    _Float16* w1t  = (_Float16*)take((size_t)128 * 128 * sizeof(_Float16));
    _Float16* w2t  = (_Float16*)take((size_t)128 * 128 * sizeof(_Float16));
    _Float16* fc1t = (_Float16*)take((size_t)256 * 128 * sizeof(_Float16));

    const int TB = 256;
    auto cdiv = [](long long a, long long b) { return (unsigned)((a + b - 1) / b); };

    // degrees + normalization terms
    k_zero_f32<<<cdiv(N, TB), TB, 0, stream>>>(deg, N);
    k_deg_count<<<cdiv(E, TB), TB, 0, stream>>>(dst, deg, E);
    k_deg_final<<<cdiv(N, TB), TB, 0, stream>>>(deg, dinv, invd, N);

    // weight / input conversion
    k_f32_to_f16<<<cdiv((long long)nF, TB), TB, 0, stream>>>(x, hf16, (int)nF);
    k_transpose_f16<<<cdiv(128 * 128, TB), TB, 0, stream>>>(W1, w1t, 128, 128);
    k_transpose_f16<<<cdiv(128 * 128, TB), TB, 0, stream>>>(W2, w2t, 128, 128);
    k_transpose_f16<<<cdiv(256 * 128, TB), TB, 0, stream>>>(fc1w, fc1t, 256, 128);

    // ---- GCN layer 1
    k_gemm_node<<<cdiv(N, 32), TB, 0, stream>>>(hf16, w1t, xw, N);
    k_zero_f32<<<cdiv((long long)nF, TB), TB, 0, stream>>>(agg, (int)nF);
    k_scatter<<<cdiv((long long)E * 32, TB), TB, 0, stream>>>(xw, src, dst, dinv, agg, E);
    k_node_update<<<cdiv((long long)nF, TB), TB, 0, stream>>>(agg, xw, invd, b1, hf16, N);

    // ---- GCN layer 2
    k_gemm_node<<<cdiv(N, 32), TB, 0, stream>>>(hf16, w2t, xw, N);
    k_zero_f32<<<cdiv((long long)nF, TB), TB, 0, stream>>>(agg, (int)nF);
    k_scatter<<<cdiv((long long)E * 32, TB), TB, 0, stream>>>(xw, src, dst, dinv, agg, E);
    k_node_update<<<cdiv((long long)nF, TB), TB, 0, stream>>>(agg, xw, invd, b2, hf16, N);

    // ---- edge head (dominant: 105 GFLOP through WMMA f16)
    k_edge_head<<<cdiv(E, 32), TB, 0, stream>>>(hf16, src, dst, fc1t, fc1b, fc2w, fc2b, out, E);
}
